// DecoderDefault_249108103487
// MI455X (gfx1250) — matrix operand order repaired and checked by reading, not compiled, not run on hardware
//
#include <hip/hip_runtime.h>
#include <math.h>
#include <stdint.h>

// ---------------------------------------------------------------------------
// HyperLSTM decoder for MI455X (gfx1250, wave32, WMMA + TDM).
// Recurrent GEMMs: v_wmma_f32_16x16x32_f16, f16 weights pre-packed into
// B-transposed (N x K) layout. A operand staged into double-buffered LDS by
// the Tensor Data Mover (tensor_load_to_lds / TENSORcnt), overlapping the DMA
// with the WMMA burst. Each wave holds all 8 M-tile accumulators so the B
// stream is read exactly once per step; 8 A fragments are preloaded into
// distinct registers so the 8 WMMAs issue back-to-back without WAR NOPs.
// ---------------------------------------------------------------------------

typedef __attribute__((ext_vector_type(16))) _Float16 v16h;
typedef __attribute__((ext_vector_type(8)))  _Float16 v8h;
typedef __attribute__((ext_vector_type(8)))  float    v8f;
typedef __attribute__((ext_vector_type(4)))  unsigned u32x4;
typedef __attribute__((ext_vector_type(8)))  int      i32x8;
typedef __attribute__((ext_vector_type(4)))  int      i32x4;

#define B_    128
#define T_    64
#define N_    1024
#define H_    256
#define XPAD  192            // x width 133 padded to multiple of KC
#define KA    1472           // A_step row: [x 192 | h 1024 | hhid 256]
#define GO_LD 9216           // gemmOut row: [xh 4096 | hh 4096 | hyp 1024]
#define KC    64             // K staging chunk (128 rows x 64 halfs = 16KB)

#if defined(__has_builtin)
#if __has_builtin(__builtin_amdgcn_tensor_load_to_lds)
#define HAVE_TDM 1
#endif
#endif

__device__ __forceinline__ float sigmf(float x) { return 1.f / (1.f + __expf(-x)); }

template <int NT>
__device__ __forceinline__ float block_sum(float v, float* red) {
  const int j = threadIdx.x;
  red[j] = v;
  __syncthreads();
#pragma unroll
  for (int s = NT / 2; s > 0; s >>= 1) {
    if (j < s) red[j] += red[j + s];
    __syncthreads();
  }
  float r = red[0];
  __syncthreads();
  return r;
}

#ifdef HAVE_TDM
// Issue one Tensor-DMA: 2D tile (KC halfs x 128 rows) of A -> LDS at lds_addr.
__device__ __forceinline__ void tdm_stage(const _Float16* gsrc, unsigned lds_addr, int lda) {
  unsigned long long ga = (unsigned long long)(uintptr_t)gsrc;
  u32x4 g0;
  g0[0] = 1u;                                              // count=1 (valid D#)
  g0[1] = lds_addr;                                        // lds_addr (bytes)
  g0[2] = (unsigned)(ga & 0xFFFFFFFFu);                    // global_addr[31:0]
  g0[3] = (unsigned)((ga >> 32) & 0x01FFFFFFu) | (2u << 30); // addr[56:32]|type=2
  unsigned td0 = (unsigned)lda;                            // tensor_dim0 (elems)
  unsigned td1 = 128u;                                     // tensor_dim1 (rows)
  unsigned long long s0 = (unsigned long long)lda;         // dim0 stride (elems)
  i32x8 g1;
  g1[0] = (int)(1u << 16);                                 // data_size=2B, mask=0
  g1[1] = (int)((td0 & 0xFFFFu) << 16);                    // td0[15:0] @ 63:48
  g1[2] = (int)(((td0 >> 16) & 0xFFFFu) | ((td1 & 0xFFFFu) << 16));
  g1[3] = (int)(((td1 >> 16) & 0xFFFFu) | ((unsigned)KC << 16)); // tile_dim0=KC
  g1[4] = 128;                                             // tile_dim1 = 128 rows
  g1[5] = (int)(s0 & 0xFFFFFFFFu);                         // stride0[31:0]
  g1[6] = (int)((s0 >> 32) & 0xFFFFu);                     // stride0[47:32]
  g1[7] = 0;
  i32x4 zz = (i32x4){0, 0, 0, 0};
#if defined(__clang_major__) && (__clang_major__ >= 23)
  i32x8 z8 = (i32x8){0, 0, 0, 0, 0, 0, 0, 0};
  __builtin_amdgcn_tensor_load_to_lds(g0, g1, zz, zz, z8, 0);
#else
  __builtin_amdgcn_tensor_load_to_lds(g0, g1, zz, zz, 0);
#endif
}
#else
__device__ __forceinline__ void coop_stage(const _Float16* gsrc, _Float16* dst,
                                           int lda, int tid) {
  for (int i = tid; i < (128 * KC) / 8; i += 128) {
    int row = (i * 8) / KC, col = (i * 8) % KC;
    *(v8h*)&dst[row * KC + col] = *(const v8h*)(gsrc + (size_t)row * lda + col);
  }
}
#endif

// ---------------------------------------------------------------------------
// WMMA GEMM with double-buffered TDM-staged A operand.
//   C[M x N] (f32) = A[M x K] (f16, row major, lda) x Bt[N x K] (f16, ldb)
// Block = 128 threads (4 waves); computes full 128-row M panel x 64 columns.
// K must be a multiple of KC. Per chunk: wait TENSORcnt for current buffer,
// barrier, kick the TDM for the next chunk into the other buffer, then run
// 2 k-steps x 8 back-to-back WMMAs with A fragments preloaded from LDS.
// ---------------------------------------------------------------------------
__global__ void wmma_gemm(const _Float16* __restrict__ A,
                          const _Float16* __restrict__ Bt,
                          float* __restrict__ C,
                          int lda, int ldb, int ldc, int K, int colOff) {
  __shared__ _Float16 shA[2][128 * KC];
  const int tid   = threadIdx.x;
  const int lane  = tid & 31;
  const int wave  = tid >> 5;
  const int nTile = blockIdx.x * 4 + wave;
  const int mbase = blockIdx.y * 128;
  const int ln15  = lane & 15;
  const int kh    = lane >> 4;

  const _Float16* brow  = Bt + (size_t)(nTile * 16 + ln15) * ldb + kh * 16;
  const _Float16* apanel = A + (size_t)mbase * lda;

  v8f acc[8] = {};
  const int nChunks = K / KC;

#ifdef HAVE_TDM
  if (tid < 32) tdm_stage(apanel, (unsigned)(uintptr_t)(void*)&shA[0][0], lda);
#else
  coop_stage(apanel, &shA[0][0], lda, tid);
#endif

  for (int ci = 0; ci < nChunks; ++ci) {
    const int k0 = ci * KC;
    const _Float16* cur = &shA[ci & 1][0];
#ifdef HAVE_TDM
    if (tid < 32) __builtin_amdgcn_s_wait_tensorcnt(0);
#endif
    __syncthreads();
    if (ci + 1 < nChunks) {               // overlap next DMA with compute
#ifdef HAVE_TDM
      if (tid < 32)
        tdm_stage(apanel + (size_t)(k0 + KC),
                  (unsigned)(uintptr_t)(void*)&shA[(ci + 1) & 1][0], lda);
#else
      coop_stage(apanel + (size_t)(k0 + KC), &shA[(ci + 1) & 1][0], lda, tid);
#endif
    }
#pragma unroll
    for (int ks = 0; ks < KC; ks += 32) {
      v16h b = *(const v16h*)(brow + k0 + ks);
      __builtin_prefetch(brow + k0 + ks + 128, 0, 1);     // global_prefetch_b8
      v16h af[8];
#pragma unroll
      for (int mt = 0; mt < 8; ++mt) {                    // batch ds_load_b128
        const _Float16* ar = &cur[(mt * 16 + ln15) * KC + ks + kh * 16];
        v8h a0 = *(const v8h*)ar;
        v8h a1 = *(const v8h*)(ar + 8);
#pragma unroll
        for (int i = 0; i < 8; ++i) { af[mt][i] = a0[i]; af[mt][i + 8] = a1[i]; }
      }
#pragma unroll
      for (int mt = 0; mt < 8; ++mt)                      // WMMA burst
        acc[mt] = __builtin_amdgcn_wmma_f32_16x16x32_f16(
            false, af[mt], false, b, (short)0, acc[mt], false, false);
    }
    __syncthreads();
  }
#pragma unroll
  for (int mt = 0; mt < 8; ++mt) {
    float* crow = C + (size_t)(mbase + mt * 16 + kh * 8) * ldc
                + colOff + nTile * 16 + ln15;
#pragma unroll
    for (int r = 0; r < 8; ++r) crow[(size_t)r * ldc] = acc[mt][r];
  }
}

// ------------------------- one-time weight packing -------------------------

__global__ void pack_bt_xh(const float* __restrict__ W_xh, _Float16* __restrict__ Bt) {
  int idx = blockIdx.x * 256 + threadIdx.x;           // [4096][192]
  if (idx >= 4096 * XPAD) return;
  int n = idx / XPAD, k = idx % XPAD;
  Bt[idx] = (k < 133) ? (_Float16)W_xh[(size_t)k * 4096 + n] : (_Float16)0.f;
}

__global__ void pack_bt_hh(const float* __restrict__ W_hh, _Float16* __restrict__ Bt) {
  int idx = blockIdx.x * 256 + threadIdx.x;           // [4096][1024]
  if (idx >= 4096 * 1024) return;
  int n = idx >> 10, k = idx & 1023;
  Bt[idx] = (_Float16)W_hh[(size_t)k * 4096 + n];
}

__global__ void pack_bt_hyp(const float* __restrict__ hyp_W, _Float16* __restrict__ Bt) {
  int idx = blockIdx.x * 256 + threadIdx.x;           // [1024][1472]
  if (idx >= 1024 * KA) return;
  int n = idx / KA, k = idx % KA;
  float v = 0.f;
  if (k < 133)                          v = hyp_W[(size_t)k * 1024 + n];                  // x
  else if (k >= XPAD && k < XPAD + N_)  v = hyp_W[(size_t)(133 + (k - XPAD)) * 1024 + n]; // h
  else if (k >= XPAD + N_)              v = hyp_W[(size_t)(1157 + (k - XPAD - N_)) * 1024 + n]; // hhid
  Bt[idx] = (_Float16)v;
}

__global__ void pack_bt_out(const float* __restrict__ W_out, _Float16* __restrict__ Bt) {
  int idx = blockIdx.x * 256 + threadIdx.x;           // [128][1024], N 123->128
  if (idx >= 128 * 1024) return;
  int n = idx >> 10, k = idx & 1023;
  Bt[idx] = (n < 123) ? (_Float16)W_out[(size_t)k * 123 + n] : (_Float16)0.f;
}

__global__ void build_decA(const float* __restrict__ sketch, const float* __restrict__ z,
                           _Float16* __restrict__ decA) {
  int idx = blockIdx.x * 256 + threadIdx.x;           // [B*T][192], row = b*T + t
  if (idx >= B_ * T_ * XPAD) return;
  int r = idx / XPAD, j = idx % XPAD;
  int b = r / T_, t = r % T_;
  float v = 0.f;
  if (j < 5)        v = sketch[((size_t)b * T_ + t) * 5 + j];
  else if (j < 133) v = z[(size_t)b * 128 + (j - 5)];
  decA[idx] = (_Float16)v;
}

__global__ void init_state(const float* __restrict__ init, const _Float16* __restrict__ decA,
                           float* __restrict__ c_state, float* __restrict__ hc_state,
                           _Float16* __restrict__ A_step) {
  int idx = blockIdx.x * 256 + threadIdx.x;
  if (idx >= B_ * KA) return;
  int b = idx / KA, col = idx % KA;
  if (col < XPAD) {
    A_step[idx] = decA[(size_t)b * T_ * XPAD + col];                    // x_0
  } else if (col < XPAD + N_) {
    int n = col - XPAD;
    A_step[idx] = (_Float16)init[(size_t)(B_ + b) * 1280 + n];          // h0
    c_state[(size_t)b * N_ + n] = init[(size_t)b * 1280 + n];           // c0
  } else {
    int j2 = col - (XPAD + N_);
    A_step[idx] = (_Float16)init[(size_t)(B_ + b) * 1280 + 1024 + j2];  // hhid0
    hc_state[(size_t)b * H_ + j2] = init[(size_t)b * 1280 + 1024 + j2]; // hc0
  }
}

// ------------------- per-step hyper cell + hyper embeddings ----------------

__global__ void hyper_step(const float* __restrict__ gemmOut,
                           float* __restrict__ hc_state,
                           float* __restrict__ zwbuf,
                           _Float16* __restrict__ A_step,
                           const float* __restrict__ hyp_ln_gamma,
                           const float* __restrict__ hyp_ln_beta,
                           const float* __restrict__ hyp_lnc_gamma,
                           const float* __restrict__ hyp_lnc_beta,
                           const float* __restrict__ W_zw,
                           const float* __restrict__ b_zw,
                           const float* __restrict__ W_zb) {
  __shared__ float red[H_];
  __shared__ float ho_sh[H_];
  const int b = blockIdx.x, j = threadIdx.x;
  const float* pre = gemmOut + (size_t)b * GO_LD + 8192;   // hyper pre-activations
  float v[4];
#pragma unroll
  for (int g = 0; g < 4; ++g) v[g] = pre[g * H_ + j];
#pragma unroll
  for (int g = 0; g < 4; ++g) {
    float m   = block_sum<H_>(v[g], red) * (1.f / H_);
    float d   = v[g] - m;
    float var = block_sum<H_>(d * d, red) * (1.f / H_);
    v[g] = hyp_ln_gamma[g * H_ + j] * (d * rsqrtf(var + 1e-3f)) + hyp_ln_beta[g * H_ + j];
  }
  float hc  = hc_state[(size_t)b * H_ + j];
  float nhc = hc * sigmf(v[2] + 1.f) + sigmf(v[0]) * tanhf(v[1]);
  hc_state[(size_t)b * H_ + j] = nhc;
  float m   = block_sum<H_>(nhc, red) * (1.f / H_);
  float d   = nhc - m;
  float var = block_sum<H_>(d * d, red) * (1.f / H_);
  float hout = tanhf(hyp_lnc_gamma[j] * (d * rsqrtf(var + 1e-3f)) + hyp_lnc_beta[j]) * sigmf(v[3]);
  ho_sh[j] = hout;
  A_step[(size_t)b * KA + (XPAD + N_) + j] = (_Float16)hout;   // hhid for t+1
  __syncthreads();
  // zw[b,8,64] (+bias) and zb[b,4,64]: 768 dot-products over H=256
#pragma unroll
  for (int q = 0; q < 3; ++q) {
    int idx = j * 3 + q;
    float s;
    if (idx < 512) {
      int g = idx >> 6, e = idx & 63;
      s = b_zw[g * 64 + e];
      const float* w = W_zw + (size_t)g * H_ * 64 + e;
      for (int h = 0; h < H_; ++h) s += ho_sh[h] * w[(size_t)h * 64];
    } else {
      int i2 = idx - 512, g = i2 >> 6, e = i2 & 63;
      s = 0.f;
      const float* w = W_zb + (size_t)g * H_ * 64 + e;
      for (int h = 0; h < H_; ++h) s += ho_sh[h] * w[(size_t)h * 64];
    }
    zwbuf[(size_t)b * 768 + idx] = s;
  }
}

// ------------- per-step gate modulation + main cell + layernorms -----------

__global__ void gates_step(const float* __restrict__ gemmOut,
                           const float* __restrict__ zwbuf,
                           float* __restrict__ c_state,
                           _Float16* __restrict__ A_step,
                           _Float16* __restrict__ outs_h,
                           const _Float16* __restrict__ decA,
                           const float* __restrict__ W_alpha,
                           const float* __restrict__ W_beta,
                           const float* __restrict__ bias,
                           const float* __restrict__ ln_all_gamma,
                           const float* __restrict__ ln_all_beta,
                           const float* __restrict__ ln_c_gamma,
                           const float* __restrict__ ln_c_beta,
                           int t) {
  __shared__ float red[N_];
  __shared__ float zw[768];
  const int b = blockIdx.x, n = threadIdx.x;
  if (n < 768) zw[n] = zwbuf[(size_t)b * 768 + n];
  __syncthreads();
  float al[8];
#pragma unroll
  for (int g = 0; g < 8; ++g) {
    float s = 0.f;
    const float* w = W_alpha + (size_t)g * 64 * N_ + n;
#pragma unroll 4
    for (int e = 0; e < 64; ++e) s += zw[g * 64 + e] * w[(size_t)e * N_];
    al[g] = s;
  }
  float be[4];
#pragma unroll
  for (int g = 0; g < 4; ++g) {
    float s = 0.f;
    const float* w = W_beta + (size_t)g * 64 * N_ + n;
#pragma unroll 4
    for (int e = 0; e < 64; ++e) s += zw[512 + g * 64 + e] * w[(size_t)e * N_];
    be[g] = s;
  }
  const float* go = gemmOut + (size_t)b * GO_LD;
  float gate[4];
#pragma unroll
  for (int g = 0; g < 4; ++g)
    gate[g] = al[g] * go[g * N_ + n] + al[4 + g] * go[4096 + g * N_ + n]
            + bias[g * N_ + n] + be[g];
#pragma unroll
  for (int g = 0; g < 4; ++g) {
    float m   = block_sum<N_>(gate[g], red) * (1.f / N_);
    float d   = gate[g] - m;
    float var = block_sum<N_>(d * d, red) * (1.f / N_);
    gate[g] = ln_all_gamma[g * N_ + n] * (d * rsqrtf(var + 1e-3f)) + ln_all_beta[g * N_ + n];
  }
  float c  = c_state[(size_t)b * N_ + n];
  float nc = c * sigmf(gate[2] + 1.f) + sigmf(gate[0]) * tanhf(gate[1]);
  c_state[(size_t)b * N_ + n] = nc;
  float m   = block_sum<N_>(nc, red) * (1.f / N_);
  float d   = nc - m;
  float var = block_sum<N_>(d * d, red) * (1.f / N_);
  float nh = tanhf(ln_c_gamma[n] * (d * rsqrtf(var + 1e-3f)) + ln_c_beta[n]) * sigmf(gate[3]);
  outs_h[((size_t)b * T_ + t) * N_ + n] = (_Float16)nh;     // row b*T+t for output GEMM
  A_step[(size_t)b * KA + XPAD + n] = (_Float16)nh;         // h for t+1
  if (n < XPAD)                                             // stage x_{t+1}
    A_step[(size_t)b * KA + n] =
        (t + 1 < T_) ? decA[((size_t)b * T_ + (t + 1)) * XPAD + n] : (_Float16)0.f;
}

// ---------------------------- mixture-coef epilogue ------------------------

__global__ void mixture(const float* __restrict__ raw, const float* __restrict__ b_out,
                        float* __restrict__ out) {
  int r = blockIdx.x * blockDim.x + threadIdx.x;
  if (r >= B_ * T_) return;
  const float* x = raw + (size_t)r * 128;
  float* o = out + (size_t)r * 126;
  float mx = -1e30f;
  for (int i = 0; i < 20; ++i) mx = fmaxf(mx, x[3 + i] + b_out[3 + i]);
  float s = 0.f;
  for (int i = 0; i < 20; ++i) s += __expf(x[3 + i] + b_out[3 + i] - mx);
  float inv = 1.f / s;
  for (int i = 0; i < 20; ++i) o[i]       = __expf(x[3 + i] + b_out[3 + i] - mx) * inv;
  for (int i = 0; i < 20; ++i) o[20 + i]  = x[23 + i] + b_out[23 + i];
  for (int i = 0; i < 20; ++i) o[40 + i]  = x[43 + i] + b_out[43 + i];
  for (int i = 0; i < 20; ++i) o[60 + i]  = __expf(x[63 + i] + b_out[63 + i]);
  for (int i = 0; i < 20; ++i) o[80 + i]  = __expf(x[83 + i] + b_out[83 + i]);
  for (int i = 0; i < 20; ++i) o[100 + i] = tanhf(x[103 + i] + b_out[103 + i]);
  float p0 = x[0] + b_out[0], p1 = x[1] + b_out[1], p2 = x[2] + b_out[2];
  float pm = fmaxf(p0, fmaxf(p1, p2));
  float e0 = __expf(p0 - pm), e1 = __expf(p1 - pm), e2 = __expf(p2 - pm);
  float ps = 1.f / (e0 + e1 + e2);
  o[120] = e0 * ps; o[121] = e1 * ps; o[122] = e2 * ps;
  o[123] = p0; o[124] = p1; o[125] = p2;
}

// ---------------------------------------------------------------------------

extern "C" void kernel_launch(void* const* d_in, const int* in_sizes, int n_in,
                              void* d_out, int out_size, void* d_ws, size_t ws_size,
                              hipStream_t stream) {
  const float* sketch        = (const float*)d_in[0];
  const float* z             = (const float*)d_in[1];
  const float* init          = (const float*)d_in[2];
  const float* W_xh          = (const float*)d_in[4];
  const float* W_hh          = (const float*)d_in[5];
  const float* bias          = (const float*)d_in[6];
  const float* ln_all_gamma  = (const float*)d_in[7];
  const float* ln_all_beta   = (const float*)d_in[8];
  const float* ln_c_gamma    = (const float*)d_in[9];
  const float* ln_c_beta     = (const float*)d_in[10];
  const float* hyp_W         = (const float*)d_in[11];
  const float* hyp_ln_gamma  = (const float*)d_in[12];
  const float* hyp_ln_beta   = (const float*)d_in[13];
  const float* hyp_lnc_gamma = (const float*)d_in[14];
  const float* hyp_lnc_beta  = (const float*)d_in[15];
  const float* W_zw          = (const float*)d_in[16];
  const float* b_zw          = (const float*)d_in[17];
  const float* W_alpha       = (const float*)d_in[18];
  const float* W_zb          = (const float*)d_in[19];
  const float* W_beta        = (const float*)d_in[20];
  const float* W_out         = (const float*)d_in[21];
  const float* b_out         = (const float*)d_in[22];

  char* p = (char*)d_ws;
  auto carve = [&](size_t bytes) -> char* {
    char* r = p;
    p += (bytes + 255) & ~(size_t)255;
    return r;
  };
  _Float16* Bt_xh   = (_Float16*)carve((size_t)4096 * XPAD * 2);
  _Float16* Bt_hh   = (_Float16*)carve((size_t)4096 * 1024 * 2);
  _Float16* Bt_hyp  = (_Float16*)carve((size_t)1024 * KA * 2);
  _Float16* Bt_out  = (_Float16*)carve((size_t)128 * 1024 * 2);
  _Float16* decA    = (_Float16*)carve((size_t)B_ * T_ * XPAD * 2);
  _Float16* A_step  = (_Float16*)carve((size_t)B_ * KA * 2);
  float*    gemmOut = (float*)carve((size_t)B_ * GO_LD * 4);
  float*    c_state = (float*)carve((size_t)B_ * N_ * 4);
  float*    hc_state= (float*)carve((size_t)B_ * H_ * 4);
  float*    zwbuf   = (float*)carve((size_t)B_ * 768 * 4);
  _Float16* outs_h  = (_Float16*)carve((size_t)B_ * T_ * N_ * 2);
  float*    rawOut  = (float*)carve((size_t)B_ * T_ * 128 * 4);

  // ---- prologue: pack f16 weights, build decoder inputs, init state ----
  build_decA<<<(B_ * T_ * XPAD + 255) / 256, 256, 0, stream>>>(sketch, z, decA);
  pack_bt_xh<<<(4096 * XPAD + 255) / 256, 256, 0, stream>>>(W_xh, Bt_xh);
  pack_bt_hh<<<(4096 * 1024 + 255) / 256, 256, 0, stream>>>(W_hh, Bt_hh);
  pack_bt_hyp<<<(1024 * KA + 255) / 256, 256, 0, stream>>>(hyp_W, Bt_hyp);
  pack_bt_out<<<(128 * 1024 + 255) / 256, 256, 0, stream>>>(W_out, Bt_out);
  init_state<<<(B_ * KA + 255) / 256, 256, 0, stream>>>(init, decA, c_state, hc_state, A_step);

  // ---- recurrent time loop: 3 WMMA GEMMs + 2 elementwise kernels / step ----
  for (int t = 0; t < T_; ++t) {
    // xh: [128 x 4096] = x(192) part of A_step x Bt_xh
    wmma_gemm<<<dim3(64, 1), 128, 0, stream>>>(A_step, Bt_xh, gemmOut,
                                               KA, XPAD, GO_LD, XPAD, 0);
    // hh: [128 x 4096] = h(1024) part of A_step x Bt_hh
    wmma_gemm<<<dim3(64, 1), 128, 0, stream>>>(A_step + XPAD, Bt_hh, gemmOut,
                                               KA, 1024, GO_LD, 1024, 4096);
    // hyper pre: [128 x 1024] = full A_step x Bt_hyp
    wmma_gemm<<<dim3(16, 1), 128, 0, stream>>>(A_step, Bt_hyp, gemmOut,
                                               KA, KA, GO_LD, KA, 8192);
    hyper_step<<<B_, H_, 0, stream>>>(gemmOut, hc_state, zwbuf, A_step,
                                      hyp_ln_gamma, hyp_ln_beta,
                                      hyp_lnc_gamma, hyp_lnc_beta,
                                      W_zw, b_zw, W_zb);
    gates_step<<<B_, N_, 0, stream>>>(gemmOut, zwbuf, c_state, A_step, outs_h, decA,
                                      W_alpha, W_beta, bias,
                                      ln_all_gamma, ln_all_beta,
                                      ln_c_gamma, ln_c_beta, t);
  }

  // ---- epilogue: output projection (WMMA) + mixture coefficients ----
  wmma_gemm<<<dim3(2, 64), 128, 0, stream>>>(outs_h, Bt_out, rawOut,
                                             1024, 1024, 128, 1024, 0);
  mixture<<<(B_ * T_ + 255) / 256, 256, 0, stream>>>(rawOut, b_out, (float*)d_out);
}
